// MemoryEfficientAttention_58042188038296
// MI455X (gfx1250) — matrix operand order repaired and checked by reading, not compiled, not run on hardware
//
#include <hip/hip_runtime.h>
#include <hip/hip_bf16.h>

typedef __attribute__((ext_vector_type(16))) __bf16          v16bf;
typedef __attribute__((ext_vector_type(8)))  float           v8f;
typedef __attribute__((ext_vector_type(8)))  unsigned short  u16x8;
typedef __attribute__((ext_vector_type(16))) unsigned short  u16x16;

namespace {
constexpr int S_  = 2048;
constexpr int B_  = 2;
constexpr int HQ_ = 32;
constexpr int HK_ = 8;
constexpr int D_  = 128;
constexpr int NREP = HQ_ / HK_;     // 4
constexpr int BM = 64;              // query rows per block (16 per wave)
constexpr int BN = 32;              // key cols per iteration
constexpr int KSTRf = D_ + 4;       // fp32 K tile row stride (floats), 528B rows
constexpr int VSTR = BN + 8;        // padded stride for transposed bf16 V tile
constexpr int PSTR = BN + 8;        // padded stride for per-wave bf16 P tile
}

// native float -> bf16 (fptrunc; lowers to v_cvt_*bf16_f32 on gfx1250)
__device__ __forceinline__ unsigned short f2bf(float f) {
  return __builtin_bit_cast(unsigned short, (__bf16)f);
}

__device__ __forceinline__ v16bf mk16(u16x8 lo, u16x8 hi) {
  u16x16 t;
#pragma unroll
  for (int i = 0; i < 8; ++i) { t[i] = lo[i]; t[8 + i] = hi[i]; }
  return __builtin_bit_cast(v16bf, t);
}

// 16 contiguous bf16 in LDS (two ds_load_b128)
__device__ __forceinline__ v16bf ldsfrag(const unsigned short* p) {
  u16x8 lo = *(const u16x8*)p;
  u16x8 hi = *(const u16x8*)(p + 8);
  return mk16(lo, hi);
}

// 16 contiguous fp32 in LDS -> bf16 fragment (four ds_load_b128 + cvt)
__device__ __forceinline__ v16bf cvt16(const float* p) {
  const float4 f0 = *(const float4*)(p);
  const float4 f1 = *(const float4*)(p + 4);
  const float4 f2 = *(const float4*)(p + 8);
  const float4 f3 = *(const float4*)(p + 12);
  u16x16 t;
  t[0]  = f2bf(f0.x); t[1]  = f2bf(f0.y); t[2]  = f2bf(f0.z); t[3]  = f2bf(f0.w);
  t[4]  = f2bf(f1.x); t[5]  = f2bf(f1.y); t[6]  = f2bf(f1.z); t[7]  = f2bf(f1.w);
  t[8]  = f2bf(f2.x); t[9]  = f2bf(f2.y); t[10] = f2bf(f2.z); t[11] = f2bf(f2.w);
  t[12] = f2bf(f3.x); t[13] = f2bf(f3.y); t[14] = f2bf(f3.z); t[15] = f2bf(f3.w);
  return __builtin_bit_cast(v16bf, t);
}

// eight async 16B copies: this thread's 128B slice of a K tile, global -> LDS.
// ASYNCcnt += 8 for the wave; pair with s_wait_asynccnt 0.
__device__ __forceinline__ void issue_k_async(const float* gbase, unsigned ldsoff) {
#pragma unroll
  for (int i = 0; i < 8; ++i) {
    const unsigned la = ldsoff + 16u * i;
    const unsigned long long ga = (unsigned long long)(size_t)(gbase + 4 * i);
    asm volatile("global_load_async_to_lds_b128 %0, %1, off"
                 :: "v"(la), "v"(ga) : "memory");
  }
}

__device__ __forceinline__ void wait_async0() {
  asm volatile("s_wait_asynccnt 0x0" ::: "memory");
}

__global__ __launch_bounds__(128)
void fa2_fwd_bf16(const float* __restrict__ Q, const float* __restrict__ K,
                  const float* __restrict__ V, float* __restrict__ O) {
  __shared__ __align__(16) float          sKf[2][BN * KSTRf];  // fp32 K tiles (double buffer)
  __shared__ __align__(16) unsigned short sVT[D_ * VSTR];      // bf16 V tile, [d][key]
  __shared__ __align__(16) unsigned short sP[4 * 16 * PSTR];   // per-wave bf16 P tiles

  const int tid  = threadIdx.x;
  const int lane = tid & 31;
  const int wave = tid >> 5;
  const int g    = lane >> 4;    // WMMA half-wave group
  const int ln   = lane & 15;

  const int q0 = blockIdx.x * BM;
  const int bh = blockIdx.y;
  const int b  = bh / HQ_;
  const int hq = bh % HQ_;
  const int hk = hq / NREP;

  const float scale = 0.08838834764831845f;  // 1/sqrt(128)

  // ---- Q fragments (A-layout, 16x32 bf16 per K-step), pre-scaled ----
  const int qrow = q0 + wave * 16 + ln;
  v16bf qf[4];
  {
    const float* qp = Q + (((size_t)qrow * B_ + b) * HQ_ + hq) * D_;
#pragma unroll
    for (int kk = 0; kk < 4; ++kk) {
      const float4 a0 = *(const float4*)(qp + kk * 32 + g * 16);
      const float4 a1 = *(const float4*)(qp + kk * 32 + g * 16 + 4);
      const float4 b0 = *(const float4*)(qp + kk * 32 + g * 16 + 8);
      const float4 b1 = *(const float4*)(qp + kk * 32 + g * 16 + 12);
      u16x16 t;
      t[0]  = f2bf(a0.x * scale); t[1]  = f2bf(a0.y * scale);
      t[2]  = f2bf(a0.z * scale); t[3]  = f2bf(a0.w * scale);
      t[4]  = f2bf(a1.x * scale); t[5]  = f2bf(a1.y * scale);
      t[6]  = f2bf(a1.z * scale); t[7]  = f2bf(a1.w * scale);
      t[8]  = f2bf(b0.x * scale); t[9]  = f2bf(b0.y * scale);
      t[10] = f2bf(b0.z * scale); t[11] = f2bf(b0.w * scale);
      t[12] = f2bf(b1.x * scale); t[13] = f2bf(b1.y * scale);
      t[14] = f2bf(b1.z * scale); t[15] = f2bf(b1.w * scale);
      qf[kk] = __builtin_bit_cast(v16bf, t);
    }
  }

  v8f acc[8] = {};           // 16x128 fp32 output tile (8 x 16x16 C/D tiles)
  float mrun[8], lrun[8];
#pragma unroll
  for (int r = 0; r < 8; ++r) { mrun[r] = -1e30f; lrun[r] = 0.0f; }

  const int nkt = (q0 + BM - 1) / BN + 1;   // uniform across the block

  // this thread's slice of each K tile
  const int krow  = tid >> 2;          // 0..31 key
  const int kdblk = (tid & 3) * 32;
  const size_t kvstride = (size_t)B_ * HK_ * D_;   // one seq step in K/V

  // prologue: async-copy K tile 0 into buffer 0
  issue_k_async(K + (((size_t)krow * B_ + b) * HK_ + hk) * D_ + kdblk,
                (unsigned)(size_t)&sKf[0][krow * KSTRf + kdblk]);

  for (int kt = 0; kt < nkt; ++kt) {
    const int kbase = kt * BN;
    const int cur   = kt & 1;

    wait_async0();        // my wave's share of K tile kt is in LDS
    __syncthreads();      // everyone's share is; prior iter's V/P/K reads done

    // ---- kick off async copy of K tile kt+1 into the other buffer ----
    if (kt + 1 < nkt)
      issue_k_async(K + (((size_t)(kbase + BN + krow) * B_ + b) * HK_ + hk) * D_ + kdblk,
                    (unsigned)(size_t)&sKf[cur ^ 1][krow * KSTRf + kdblk]);

    // ---- stage V tile kt transposed: [d][key] bf16, 2 keys per b32 ----
    {
      const int j0   = (tid >> 3) * 2;    // key pair
      const int dblk = (tid & 7) * 16;
      const float* v0 = V + (((size_t)(kbase + j0) * B_ + b) * HK_ + hk) * D_ + dblk;
      const float* v1 = v0 + kvstride;
#pragma unroll
      for (int i = 0; i < 4; ++i) {
        const float4 x0 = *(const float4*)(v0 + 4 * i);
        const float4 x1 = *(const float4*)(v1 + 4 * i);
        const float a0[4] = {x0.x, x0.y, x0.z, x0.w};
        const float a1[4] = {x1.x, x1.y, x1.z, x1.w};
#pragma unroll
        for (int c = 0; c < 4; ++c)
          *(unsigned*)&sVT[(dblk + 4 * i + c) * VSTR + j0] =
              (unsigned)f2bf(a0[c]) | ((unsigned)f2bf(a1[c]) << 16);
      }
      if (kbase + BN + j0 < S_)   // prefetch next V tile (global_prefetch_b8)
        __builtin_prefetch(v0 + BN * kvstride, 0, 1);
    }
    __syncthreads();

    // ---- S = Q * K^T : 2 N-tiles x 4 K-steps of v_wmma_f32_16x16x32_bf16 ----
    const float* kbuf = sKf[cur];
    v8f sc[2];
#pragma unroll
    for (int nt = 0; nt < 2; ++nt) {
      v8f c = {};
#pragma unroll
      for (int kk = 0; kk < 4; ++kk) {
        v16bf kb = cvt16(&kbuf[(nt * 16 + ln) * KSTRf + kk * 32 + g * 16]);
        c = __builtin_amdgcn_wmma_f32_16x16x32_bf16(false, qf[kk], false, kb,
                                                    (short)0, c, false, false);
      }
      sc[nt] = c;
    }

    // ---- online softmax (row m = r + 8*g lives across the 16-lane half) ----
    unsigned short* pw = &sP[wave * 16 * PSTR];
    const int qp0 = q0 + wave * 16 + 8 * g;
#pragma unroll
    for (int r = 0; r < 8; ++r) {
      const int qpos = qp0 + r;
      float s0 = sc[0][r];
      float s1 = sc[1][r];
      if (kbase + ln > qpos)      s0 = -1e30f;   // causal mask
      if (kbase + 16 + ln > qpos) s1 = -1e30f;
      float tmax = fmaxf(s0, s1);
#pragma unroll
      for (int off = 8; off >= 1; off >>= 1)
        tmax = fmaxf(tmax, __shfl_xor(tmax, off, 32));
      const float mn = fmaxf(mrun[r], tmax);
      const float al = __expf(mrun[r] - mn);
      const float p0 = __expf(s0 - mn);
      const float p1 = __expf(s1 - mn);
      float ps = p0 + p1;
#pragma unroll
      for (int off = 8; off >= 1; off >>= 1)
        ps += __shfl_xor(ps, off, 32);
      lrun[r] = lrun[r] * al + ps;
      mrun[r] = mn;
#pragma unroll
      for (int dt = 0; dt < 8; ++dt) acc[dt][r] *= al;
      // P (C-layout) -> LDS for A-layout reload
      pw[(8 * g + r) * PSTR + ln]      = f2bf(p0);
      pw[(8 * g + r) * PSTR + 16 + ln] = f2bf(p1);
    }
    __syncthreads();

    // ---- O += P * V : 8 D-tiles, one 16x16x32 WMMA each ----
    u16x8 plo = *(const u16x8*)&pw[ln * PSTR + g * 16];
    u16x8 phi = *(const u16x8*)&pw[ln * PSTR + g * 16 + 8];
    const v16bf pa = mk16(plo, phi);
#pragma unroll
    for (int dt = 0; dt < 8; ++dt) {
      v16bf vb = ldsfrag(&sVT[(dt * 16 + ln) * VSTR + g * 16]);
      acc[dt] = __builtin_amdgcn_wmma_f32_16x16x32_bf16(false, pa, false, vb,
                                                        (short)0, acc[dt], false, false);
    }
  }

  // ---- epilogue: normalize and store fp32 [S,B,HQ*D] ----
  float inv[8];
#pragma unroll
  for (int r = 0; r < 8; ++r) inv[r] = 1.0f / lrun[r];

  const int orow0 = q0 + wave * 16 + 8 * g;
#pragma unroll
  for (int r = 0; r < 8; ++r) {
    float* op = O + ((size_t)(orow0 + r) * B_ + b) * (HQ_ * D_) + hq * D_ + ln;
#pragma unroll
    for (int dt = 0; dt < 8; ++dt)
      op[dt * 16] = acc[dt][r] * inv[r];
  }
}

extern "C" void kernel_launch(void* const* d_in, const int* /*in_sizes*/, int /*n_in*/,
                              void* d_out, int /*out_size*/, void* /*d_ws*/, size_t /*ws_size*/,
                              hipStream_t stream) {
  const float* q = (const float*)d_in[0];
  const float* k = (const float*)d_in[1];
  const float* v = (const float*)d_in[2];
  float* o = (float*)d_out;
  dim3 grid(S_ / BM, B_ * HQ_);
  fa2_fwd_bf16<<<grid, dim3(128), 0, stream>>>(q, k, v, o);
}
